// DREAM_70076686402271
// MI455X (gfx1250) — matrix operand order repaired and checked by reading, not compiled, not run on hardware
//
#include <hip/hip_runtime.h>
#include <hip/hip_bf16.h>
#include <math.h>

// Problem constants (match reference)
#define B_  16
#define T_  2048
#define D_  512
#define H_  1024
#define R_  16

typedef __bf16 bf16_t;
typedef __attribute__((ext_vector_type(16))) __bf16 v16bf;
typedef __attribute__((ext_vector_type(8)))  __bf16 v8bf;
typedef __attribute__((ext_vector_type(8)))  float  v8f;

// ---------------------------------------------------------------------------
// Utility: fp32 -> bf16 bulk convert
// ---------------------------------------------------------------------------
__global__ void f32_to_bf16_kernel(const float* __restrict__ src,
                                   bf16_t* __restrict__ dst, int n) {
  for (int i = blockIdx.x * blockDim.x + threadIdx.x; i < n;
       i += gridDim.x * blockDim.x)
    dst[i] = (bf16_t)src[i];
}

// W (K x N, fp32, row-major) -> WT (N x K, bf16, row-major)
__global__ void transpose_to_bf16_kernel(const float* __restrict__ W,
                                         bf16_t* __restrict__ WT,
                                         int K, int N) {
  int total = K * N;
  for (int i = blockIdx.x * blockDim.x + threadIdx.x; i < total;
       i += gridDim.x * blockDim.x) {
    int nrow = i / K;      // output row (original column)
    int kcol = i - nrow * K;
    WT[(size_t)nrow * K + kcol] = (bf16_t)W[(size_t)kcol * N + nrow];
  }
}

// ---------------------------------------------------------------------------
// lam[row] = sigmoid(dot(x[row,:], Wd) + bd), one wave32 per row
// ---------------------------------------------------------------------------
__global__ void lam_kernel(const float* __restrict__ x,
                           const float* __restrict__ Wd,
                           const float* __restrict__ bd,
                           float* __restrict__ lam, int rows) {
  int gwave = (blockIdx.x * blockDim.x + threadIdx.x) >> 5;
  int lane  = threadIdx.x & 31;
  if (gwave >= rows) return;
  const float* xr = x + (size_t)gwave * D_;
  float p = 0.f;
  #pragma unroll
  for (int i = 0; i < D_ / 32; ++i) p += xr[lane + 32 * i] * Wd[lane + 32 * i];
  #pragma unroll
  for (int off = 16; off; off >>= 1) p += __shfl_down(p, off, 32);
  if (lane == 0) lam[gwave] = 1.f / (1.f + __expf(-(p + bd[0])));
}

// ---------------------------------------------------------------------------
// WMMA bf16 GEMM: C[M,N] = A[M,K] (bf16 row-major) @ Bt[N,K]^T (bf16 row-major)
// Block: 256 threads = 8 waves; block tile 16(M) x 128(N); wave tile 16x16.
// Fragment layouts used here (A/B: contiguous K, not the ISA table order; C/D per ISA):
//   A 16x32 bf16: lane (hf=lane/16, m=lane%16) holds row m,
//                 K = 16*hf..16*hf+15  (two 16B loads)
//   B 32x16 bf16: lane holds column n=lane%16, K = 16*hf..16*hf+15 (one 32B load
//                 from the pre-transposed weight, contiguous in K)
//   C/D 16x16 f32: VGPR j -> (M = j + 8*hf, N = lane%16)
// ---------------------------------------------------------------------------
template <int OUT_BF16>
__global__ void wmma_gemm_kernel(const bf16_t* __restrict__ A,
                                 const bf16_t* __restrict__ Bt,
                                 void* __restrict__ Cout,
                                 int M, int N, int K) {
  const int lane = threadIdx.x & 31;
  const int wave = threadIdx.x >> 5;
  const int hf   = lane >> 4;
  const int l16  = lane & 15;
  const int m0   = blockIdx.x * 16;
  const int n0   = blockIdx.y * 128 + wave * 16;
  if (n0 >= N) return;

  const bf16_t* arow = A  + (size_t)(m0 + l16) * K;
  const bf16_t* brow = Bt + (size_t)(n0 + l16) * K;

  v8f acc = {};
  for (int k0 = 0; k0 < K; k0 += 32) {
    __builtin_prefetch(arow + k0 + 128, 0, 1);
    __builtin_prefetch(brow + k0 + 128, 0, 1);
    v8bf alo = *(const v8bf*)(arow + k0 + 16 * hf);
    v8bf ahi = *(const v8bf*)(arow + k0 + 16 * hf + 8);
    v16bf a;
    #pragma unroll
    for (int i = 0; i < 8; ++i) { a[i] = alo[i]; a[i + 8] = ahi[i]; }
    v16bf b = *(const v16bf*)(brow + k0 + 16 * hf);
    acc = __builtin_amdgcn_wmma_f32_16x16x32_bf16(false, a, false, b,
                                                  (short)0, acc, false, false);
  }

  if (OUT_BF16) {
    bf16_t* C = (bf16_t*)Cout;
    #pragma unroll
    for (int j = 0; j < 8; ++j)
      C[(size_t)(m0 + j + 8 * hf) * N + n0 + l16] = (bf16_t)acc[j];
  } else {
    float* C = (float*)Cout;
    #pragma unroll
    for (int j = 0; j < 8; ++j)
      C[(size_t)(m0 + j + 8 * hf) * N + n0 + l16] = acc[j];
  }
}

// ---------------------------------------------------------------------------
// Fast-weight scan (parallel over B*H, sequential over T):
//   F[r] = lam_t*F[r] + v_t[h]*k_t[r];  read = sum_r F[r]*q_t[r]
//   u[b,t,h] += read + bias[h]          (u already holds xw from the GEMM)
// Writes F_fin at the end. Grid: B_*(H_/256) blocks of 256 threads.
// ---------------------------------------------------------------------------
__global__ void fastweight_scan_kernel(const float* __restrict__ qf,
                                       const float* __restrict__ kf,
                                       const float* __restrict__ lamp,
                                       const bf16_t* __restrict__ vbf,
                                       const float* __restrict__ bias,
                                       float* __restrict__ u,
                                       float* __restrict__ Ffin) {
  __shared__ float klds[256];
  __shared__ float qlds[256];
  __shared__ float llds[16];
  const int tid = threadIdx.x;
  const int b   = blockIdx.x >> 2;
  const int h   = ((blockIdx.x & 3) << 8) + tid;

  float F[R_];
  #pragma unroll
  for (int r = 0; r < R_; ++r) F[r] = 0.f;
  const float bh = bias[h];

  for (int t0 = 0; t0 < T_; t0 += 16) {
    __syncthreads();
    size_t base = ((size_t)b * T_ + t0) * R_;
    klds[tid] = kf[base + tid];
    qlds[tid] = qf[base + tid];
    if (tid < 16) llds[tid] = lamp[(size_t)b * T_ + t0 + tid];
    __syncthreads();
    #pragma unroll 4
    for (int s = 0; s < 16; ++s) {
      float lam_s = llds[s];
      size_t row  = (size_t)b * T_ + t0 + s;
      float vv = (float)vbf[row * H_ + h];
      float rd = 0.f;
      #pragma unroll
      for (int r = 0; r < R_; ++r) {
        F[r] = lam_s * F[r] + vv * klds[s * 16 + r];
        rd  += F[r] * qlds[s * 16 + r];
      }
      u[row * H_ + h] += rd + bh;
    }
  }
  #pragma unroll
  for (int r = 0; r < R_; ++r)
    Ffin[((size_t)b * H_ + h) * R_ + r] = F[r];
}

// ---------------------------------------------------------------------------
// Device-wide barrier (persistent cooperative kernel)
// ---------------------------------------------------------------------------
__device__ inline void grid_sync(unsigned* cnt, unsigned* gen, unsigned nwg) {
  __syncthreads();
  if (threadIdx.x == 0) {
    unsigned g = __hip_atomic_load(gen, __ATOMIC_RELAXED, __HIP_MEMORY_SCOPE_AGENT);
    unsigned arrived =
        __hip_atomic_fetch_add(cnt, 1u, __ATOMIC_ACQ_REL, __HIP_MEMORY_SCOPE_AGENT);
    if (arrived == nwg - 1u) {
      __hip_atomic_store(cnt, 0u, __ATOMIC_RELAXED, __HIP_MEMORY_SCOPE_AGENT);
      __hip_atomic_fetch_add(gen, 1u, __ATOMIC_RELEASE, __HIP_MEMORY_SCOPE_AGENT);
    } else {
      while (__hip_atomic_load(gen, __ATOMIC_ACQUIRE, __HIP_MEMORY_SCOPE_AGENT) == g)
        __builtin_amdgcn_s_sleep(1);
    }
  }
  __syncthreads();
}

// ---------------------------------------------------------------------------
// h-recurrence: h_t = tanh(u_t + h_{t-1} @ Wh), 64 persistent WGs, each owns
// a 16-column tile of h. Wht tile (1024x16 bf16 = 32KB) resident in LDS.
// Split-K over 8 waves (4 WMMAs each), LDS reduce, double-buffered bf16 state.
// ---------------------------------------------------------------------------
__global__ __launch_bounds__(256, 1)
void recurrence_kernel(const bf16_t* __restrict__ WhT,   // H x H bf16 (N-major)
                       const float*  __restrict__ u,     // (B,T,H) fp32
                       bf16_t* __restrict__ hbf,         // 2 x (16*H) bf16 state
                       float* __restrict__ hs_out,       // (B,T,H)
                       float* __restrict__ hfin_out,     // (B,H)
                       unsigned* __restrict__ bar) {
  __shared__ __align__(32) bf16_t whlds[16 * H_];   // 32 KB: Wh^T rows n0..n0+15
  __shared__ __align__(16) float  red[8 * 256];     // 8 KB: split-K partials

  const int tid  = threadIdx.x;
  const int lane = tid & 31;
  const int wave = tid >> 5;
  const int hf   = lane >> 4;
  const int l16  = lane & 15;
  const int n0   = blockIdx.x * 16;
  const unsigned nwg = gridDim.x;

  // Stage this WG's slice of Wh^T into LDS once (16 rows x 1024 bf16).
  {
    const v8bf* src = (const v8bf*)(WhT + (size_t)n0 * H_);
    v8bf* dst = (v8bf*)whlds;
    for (int i = tid; i < (16 * H_) / 8; i += 256) dst[i] = src[i];
  }
  __syncthreads();

  const int k0base = wave * 128;

  for (int t = 0; t < T_; ++t) {
    const bf16_t* hprev = hbf + (size_t)(t & 1) * (16 * H_);
    const bf16_t* arow  = hprev + (size_t)l16 * H_;

    v8f acc = {};
    #pragma unroll
    for (int kk = 0; kk < 4; ++kk) {
      const int k0 = k0base + kk * 32;
      v8bf alo = *(const v8bf*)(arow + k0 + 16 * hf);
      v8bf ahi = *(const v8bf*)(arow + k0 + 16 * hf + 8);
      v16bf a;
      #pragma unroll
      for (int i = 0; i < 8; ++i) { a[i] = alo[i]; a[i + 8] = ahi[i]; }
      v16bf b = *(const v16bf*)(whlds + l16 * H_ + k0 + 16 * hf);
      acc = __builtin_amdgcn_wmma_f32_16x16x32_bf16(false, a, false, b,
                                                    (short)0, acc, false, false);
    }

    #pragma unroll
    for (int j = 0; j < 8; ++j)
      red[wave * 256 + (j + 8 * hf) * 16 + l16] = acc[j];
    __syncthreads();

    {
      const int bb = tid >> 4;       // batch row 0..15
      const int nn = tid & 15;       // column within tile
      float s = 0.f;
      #pragma unroll
      for (int w = 0; w < 8; ++w) s += red[w * 256 + tid];
      size_t row = (size_t)bb * T_ + t;
      float val = tanhf(u[row * H_ + n0 + nn] + s);
      hs_out[row * H_ + n0 + nn] = val;
      hbf[(size_t)((t + 1) & 1) * (16 * H_) + bb * H_ + n0 + nn] = (bf16_t)val;
      if (t == T_ - 1) hfin_out[bb * H_ + n0 + nn] = val;
    }
    grid_sync(bar, bar + 1, nwg);    // also makes red[] safe to reuse
  }
}

// ---------------------------------------------------------------------------
// Host launcher
// ---------------------------------------------------------------------------
extern "C" void kernel_launch(void* const* d_in, const int* in_sizes, int n_in,
                              void* d_out, int out_size, void* d_ws, size_t ws_size,
                              hipStream_t stream) {
  const float* x    = (const float*)d_in[0];
  const float* Wx   = (const float*)d_in[1];
  const float* Wh   = (const float*)d_in[2];
  const float* Wq   = (const float*)d_in[3];
  const float* Wk   = (const float*)d_in[4];
  const float* Wv   = (const float*)d_in[5];
  const float* Wd   = (const float*)d_in[6];
  const float* bias = (const float*)d_in[7];
  const float* bd   = (const float*)d_in[8];

  const size_t MT = (size_t)B_ * T_;   // 32768 rows

  // Workspace layout
  char* ws = (char*)d_ws;
  size_t off = 0;
  auto alloc = [&](size_t bytes) -> void* {
    void* p = ws + off;
    off = (off + bytes + 255) & ~(size_t)255;
    return p;
  };
  unsigned* ctrl = (unsigned*)alloc(256);                       // barrier cnt/gen
  bf16_t* hbf    = (bf16_t*)alloc(2 * 16 * H_ * sizeof(bf16_t)); // h state x2
  const size_t zero_bytes = off;                                 // zeroed region
  bf16_t* Xbf = (bf16_t*)alloc(MT * D_ * sizeof(bf16_t));
  bf16_t* WxT = (bf16_t*)alloc((size_t)H_ * D_ * sizeof(bf16_t));
  bf16_t* WvT = (bf16_t*)alloc((size_t)H_ * D_ * sizeof(bf16_t));
  bf16_t* WqT = (bf16_t*)alloc((size_t)R_ * D_ * sizeof(bf16_t));
  bf16_t* WkT = (bf16_t*)alloc((size_t)R_ * D_ * sizeof(bf16_t));
  bf16_t* WhT = (bf16_t*)alloc((size_t)H_ * H_ * sizeof(bf16_t));
  float*  qf  = (float*)alloc(MT * R_ * sizeof(float));
  float*  kf  = (float*)alloc(MT * R_ * sizeof(float));
  float*  lam = (float*)alloc(MT * sizeof(float));
  bf16_t* vbf = (bf16_t*)alloc(MT * H_ * sizeof(bf16_t));
  float*  u   = (float*)alloc(MT * H_ * sizeof(float));

  // Outputs: hs (B,T,H), h_fin (B,H), F_fin (B,H,R) — fp32, concatenated.
  float* hs   = (float*)d_out;
  float* hfin = hs + MT * H_;
  float* Ffin = hfin + (size_t)B_ * H_;

  // Zero barrier counters + initial h state (graph-capture legal).
  hipMemsetAsync(d_ws, 0, zero_bytes, stream);

  // 1) Convert x to bf16; transpose-convert weights to N-major bf16.
  f32_to_bf16_kernel<<<8192, 256, 0, stream>>>(x, Xbf, (int)(MT * D_));
  transpose_to_bf16_kernel<<<2048, 256, 0, stream>>>(Wx, WxT, D_, H_);
  transpose_to_bf16_kernel<<<2048, 256, 0, stream>>>(Wv, WvT, D_, H_);
  transpose_to_bf16_kernel<<<64,   256, 0, stream>>>(Wq, WqT, D_, R_);
  transpose_to_bf16_kernel<<<64,   256, 0, stream>>>(Wk, WkT, D_, R_);
  transpose_to_bf16_kernel<<<4096, 256, 0, stream>>>(Wh, WhT, H_, H_);

  // 2) lambda = sigmoid(x @ Wd + bd)
  lam_kernel<<<(int)(MT / 8), 256, 0, stream>>>(x, Wd, bd, lam, (int)MT);

  // 3) Time-parallel projections via WMMA.
  {
    dim3 gH((unsigned)(MT / 16), H_ / 128);
    dim3 gR((unsigned)(MT / 16), 1);
    wmma_gemm_kernel<0><<<gH, 256, 0, stream>>>(Xbf, WxT, u,   (int)MT, H_, D_); // xw -> u
    wmma_gemm_kernel<1><<<gH, 256, 0, stream>>>(Xbf, WvT, vbf, (int)MT, H_, D_); // v (bf16)
    wmma_gemm_kernel<0><<<gR, 256, 0, stream>>>(Xbf, WqT, qf,  (int)MT, R_, D_); // q
    wmma_gemm_kernel<0><<<gR, 256, 0, stream>>>(Xbf, WkT, kf,  (int)MT, R_, D_); // k
  }

  // 4) Fast-weight scan: u += read + bias; emits F_fin.
  fastweight_scan_kernel<<<B_ * (H_ / 256), 256, 0, stream>>>(qf, kf, lam, vbf,
                                                              bias, u, Ffin);

  // 5) Sequential h-recurrence (persistent cooperative WMMA kernel).
  recurrence_kernel<<<H_ / 16, 256, 0, stream>>>(WhT, u, hbf, hs, hfin, ctrl);
}
